// GraphUpdate_4750233829573
// MI455X (gfx1250) — compile-verified, matrix operand order checked by reading, not run on hardware
//
#include <hip/hip_runtime.h>

typedef __attribute__((ext_vector_type(16))) _Float16 v16h;
typedef __attribute__((ext_vector_type(8)))  _Float16 v8h;
typedef __attribute__((ext_vector_type(8)))  float    v8f;
typedef unsigned int u32;

#define DEV __device__ __forceinline__

// problem dims
constexpr int MD = 64, ND = 64, DD = 64;            // M, N, feature dim
constexpr long long RR = (long long)MD * ND * ND;   // 262144 edge rows

// ---- workspace layout ----
// f16 weights, stored FRAGMENT-SWIZZLED (offsets in halves)
constexpr int HOFF_WQ   = 0;
constexpr int HOFF_WK   = 4096;
constexpr int HOFF_PW1  = 8192;
constexpr int HOFF_PW2  = 12288;
constexpr int HOFF_F2W1 = 16384;
constexpr int HOFF_F2W2 = 20480;
constexpr int HOFF_FW1  = 24576;  // 128x64
constexpr int HOFF_FW2  = 32768;
constexpr int HOFF_GW1  = 36864;  // 128x64
constexpr int HOFF_GW2  = 45056;
// big buffers (byte offsets)
constexpr size_t EDGE_H   = (size_t)RR * 64;              // halves per edge tensor
constexpr size_t B_XIJH   = 131072;
constexpr size_t B_XQT    = B_XIJH + EDGE_H * 2;
constexpr size_t B_XKT    = B_XQT  + EDGE_H * 2;
constexpr size_t B_PHI    = B_XKT  + EDGE_H * 2;
constexpr size_t B_SCO    = B_PHI  + EDGE_H * 2;
constexpr size_t B_MSG    = B_SCO  + EDGE_H * 2;
constexpr size_t B_AGG    = B_MSG  + EDGE_H * 2;          // M*N*64 f32

constexpr int OUT_EDGE_BASE = 64 * 64 * 96;               // node region is first

// ---------------- async global->LDS copy (CDNA5, ASYNCcnt) ----------------
// Low 32 bits of a flat shared-space address are the LDS byte offset
// (ISA 10.2: LDS aperture address = addr[31:0]).
DEV void async_cp16(void* lds, const void* g) {
  u32 laddr = (u32)(uintptr_t)lds;
  asm volatile("global_load_async_to_lds_b128 %0, %1, off"
               :: "v"(laddr), "v"(g) : "memory");
}
DEV void async_wait_all() {
  asm volatile("s_wait_asynccnt 0x0" ::: "memory");
}
// copy n 16-byte chunks, contiguous global -> contiguous LDS
DEV void async_copy(void* lds, const void* g, int nchunks, int tid, int nthr) {
  char* l = (char*)lds;
  const char* s = (const char*)g;
  for (int i = tid; i < nchunks; i += nthr)
    async_cp16(l + (size_t)i * 16, s + (size_t)i * 16);
}

// ---------------- WMMA fragment helpers ----------------------------------
// A fragment (16x32 f16, ISA 7.12.2): per lane two contiguous 16B runs.
DEV v16h load_frag_a(const _Float16* p0, int ld, int lane) {
  const _Float16* p = p0 + (lane & 15) * ld + ((lane >> 4) & 1) * 8;
  v8h lo = *(const v8h*)p;
  v8h hh = *(const v8h*)(p + 16);
  v16h a;
#pragma unroll
  for (int e = 0; e < 8; ++e) { a[e] = lo[e]; a[8 + e] = hh[e]; }
  return a;
}

// B fragment from fragment-swizzled storage: 16 contiguous halves per lane.
// Tile (kt,nt) of a Kx64 matrix lives at base + (kt*4+nt)*512 halves.
DEV v16h load_frag_b_swz(const _Float16* base, int lane) {
  return *(const v16h*)(base + lane * 16);
}

DEV void zero_acc(v8f acc[4]) {
#pragma unroll
  for (int nt = 0; nt < 4; ++nt)
#pragma unroll
    for (int v = 0; v < 8; ++v) acc[nt][v] = 0.0f;
}

// one wave computes a 16x64 strip (4 tiles) of a 64x64 output.
// A: row-major [rows x KD] f16 in LDS.  Bs: fragment-swizzled [KD x 64] in LDS.
template <int KD>
DEV void gemm_strip(const _Float16* A, int lda, const _Float16* Bs,
                    int mt, int lane, v8f acc[4]) {
#pragma unroll
  for (int kt = 0; kt < KD / 32; ++kt) {
    v16h a = load_frag_a(A + (mt * 16) * lda + kt * 32, lda, lane);
#pragma unroll
    for (int nt = 0; nt < 4; ++nt) {
      v16h b = load_frag_b_swz(Bs + ((kt * 4 + nt) << 9), lane);
      acc[nt] = __builtin_amdgcn_wmma_f32_16x16x32_f16(
          false, a, false, b, (short)0, acc[nt], false, false);
    }
  }
}

// ---------------- weight conversion + swizzle -----------------------------
// dst element i -> tile=i>>9, lane=(i>>4)&31, e=i&15 ; src row-major [K x 64]
__global__ void __launch_bounds__(256)
prep_weights(const float* wq, const float* wk, const float* pw1, const float* pw2,
             const float* f2w1, const float* f2w2, const float* fw1, const float* fw2,
             const float* gw1, const float* gw2, _Float16* wsw) {
  const float* src = nullptr; _Float16* dst = nullptr; int n = 0;
  switch (blockIdx.x) {
    case 0: src = wq;   dst = wsw + HOFF_WQ;   n = 4096; break;
    case 1: src = wk;   dst = wsw + HOFF_WK;   n = 4096; break;
    case 2: src = pw1;  dst = wsw + HOFF_PW1;  n = 4096; break;
    case 3: src = pw2;  dst = wsw + HOFF_PW2;  n = 4096; break;
    case 4: src = f2w1; dst = wsw + HOFF_F2W1; n = 4096; break;
    case 5: src = f2w2; dst = wsw + HOFF_F2W2; n = 4096; break;
    case 6: src = fw1;  dst = wsw + HOFF_FW1;  n = 8192; break;
    case 7: src = fw2;  dst = wsw + HOFF_FW2;  n = 4096; break;
    case 8: src = gw1;  dst = wsw + HOFF_GW1;  n = 8192; break;
    case 9: src = gw2;  dst = wsw + HOFF_GW2;  n = 4096; break;
  }
  for (int i = threadIdx.x; i < n; i += 256) {
    int tile = i >> 9, lane = (i >> 4) & 31, e = i & 15;
    int hi = lane >> 4, nc = lane & 15;
    int kt = tile >> 2, nt = tile & 3;
    int kl = (e < 8) ? (e + 8 * hi) : (8 + e + 8 * hi);
    dst[i] = (_Float16)src[(kt * 32 + kl) * 64 + nt * 16 + nc];
  }
}

// ---------------- stage A: q/k projections + phi MLP ----------------------
__global__ void __launch_bounds__(128)
k_edge_pre(const float* __restrict__ xij, const _Float16* __restrict__ wsw,
           const float* __restrict__ phi_b1, const float* __restrict__ phi_b2,
           _Float16* __restrict__ xij_h, _Float16* __restrict__ xqT,
           _Float16* __restrict__ xkT, _Float16* __restrict__ phi_h) {
  __shared__ alignas(32) _Float16 Xh[64 * 64];
  __shared__ alignas(32) _Float16 Wq[64 * 64], Wk[64 * 64], P1[64 * 64], P2[64 * 64];
  __shared__ alignas(32) _Float16 stage[64 * 64];
  const int tid = threadIdx.x, lane = tid & 31, wv = tid >> 5;
  const int bx = blockIdx.x, m = bx >> 6, n1 = bx & 63;
  const size_t slab = (size_t)bx * 4096;

  // weights -> LDS via async DMA (already swizzled in ws)
  async_copy(Wq, wsw + HOFF_WQ,  512, tid, 128);
  async_copy(Wk, wsw + HOFF_WK,  512, tid, 128);
  async_copy(P1, wsw + HOFF_PW1, 512, tid, 128);
  async_copy(P2, wsw + HOFF_PW2, 512, tid, 128);

  __builtin_prefetch(xij + slab + 4096, 0, 0);       // global_prefetch_b8
  for (int i = tid; i < 4096; i += 128) {            // xij f32 -> f16
    _Float16 h = (_Float16)xij[slab + i];
    Xh[i] = h;
    xij_h[slab + i] = h;
  }
  async_wait_all();
  __syncthreads();

  const int hi = (lane >> 4) & 1, nn = lane & 15;
  v8f acc[4];

  // ---- Q = X @ Wq, store transposed [k][n2] then channel-major to global
  zero_acc(acc);
  gemm_strip<64>(Xh, 64, Wq, wv, lane, acc);
#pragma unroll
  for (int nt = 0; nt < 4; ++nt)
#pragma unroll
    for (int v = 0; v < 8; ++v) {
      int row = wv * 16 + v + 8 * hi, col = nt * 16 + nn;
      stage[col * 64 + row] = (_Float16)acc[nt][v];
    }
  __syncthreads();
  {
    const u32* s = (const u32*)stage;
    u32* g = (u32*)xqT;
    for (int idx = tid; idx < 2048; idx += 128) {
      int k = idx >> 5, dw = idx & 31;
      g[(((size_t)m * 64 + k) * 64 + n1) * 32 + dw] = s[idx];
    }
  }
  __syncthreads();

  // ---- K = X @ Wk
  zero_acc(acc);
  gemm_strip<64>(Xh, 64, Wk, wv, lane, acc);
#pragma unroll
  for (int nt = 0; nt < 4; ++nt)
#pragma unroll
    for (int v = 0; v < 8; ++v) {
      int row = wv * 16 + v + 8 * hi, col = nt * 16 + nn;
      stage[col * 64 + row] = (_Float16)acc[nt][v];
    }
  __syncthreads();
  {
    const u32* s = (const u32*)stage;
    u32* g = (u32*)xkT;
    for (int idx = tid; idx < 2048; idx += 128) {
      int k = idx >> 5, dw = idx & 31;
      g[(((size_t)m * 64 + k) * 64 + n1) * 32 + dw] = s[idx];
    }
  }
  __syncthreads();

  // ---- phi layer 1: tanh(X @ P1 + b1) -> stage (row-major)
  zero_acc(acc);
  gemm_strip<64>(Xh, 64, P1, wv, lane, acc);
#pragma unroll
  for (int nt = 0; nt < 4; ++nt) {
    float b = phi_b1[nt * 16 + nn];
#pragma unroll
    for (int v = 0; v < 8; ++v) {
      int row = wv * 16 + v + 8 * hi, col = nt * 16 + nn;
      stage[row * 64 + col] = (_Float16)tanhf(acc[nt][v] + b);
    }
  }
  __syncthreads();

  // ---- phi layer 2 -> phi_h (row-major f16, global)
  zero_acc(acc);
  gemm_strip<64>(stage, 64, P2, wv, lane, acc);
#pragma unroll
  for (int nt = 0; nt < 4; ++nt) {
    float b = phi_b2[nt * 16 + nn];
#pragma unroll
    for (int v = 0; v < 8; ++v) {
      int row = wv * 16 + v + 8 * hi, col = nt * 16 + nn;
      phi_h[slab + row * 64 + col] = (_Float16)(acc[nt][v] + b);
    }
  }
}

// ---------------- stage B: per-channel attention einsum -------------------
__global__ void __launch_bounds__(128)
k_scores(const _Float16* __restrict__ xqT, const _Float16* __restrict__ xkT,
         _Float16* __restrict__ scores_h) {
  __shared__ alignas(32) _Float16 Q[64 * 64];
  __shared__ alignas(32) _Float16 Kswz[64 * 64];   // fragment-swizzled K_k
  const int tid = threadIdx.x, lane = tid & 31, wv = tid >> 5;
  const int bx = blockIdx.x, m = bx >> 6, k = bx & 63;
  const size_t slab = (size_t)bx * 4096;

  async_copy(Q, xqT + slab, 512, tid, 128);          // Q: plain async DMA
  __builtin_prefetch(xkT + slab + 4096, 0, 0);
  {
    const u32* ks = (const u32*)(xkT + slab);
    for (int idx = tid; idx < 2048; idx += 128) {
      // swizzle K element pair (j, l..l+1) into B-fragment layout
      u32 d = ks[idx];
      union { u32 u; _Float16 h[2]; } cv; cv.u = d;
      int j = idx >> 5, l = (idx & 31) * 2;
      int kt = j >> 5, kl = j & 31;
      int fhi = (kl >> 3) & 1;
      int e = (kl & 7) + ((kl >> 4) << 3);
      int nt = l >> 4, nc = l & 15;
      _Float16* base = Kswz + ((kt * 4 + nt) << 9) + e;
      base[(fhi * 16 + nc) * 16]     = cv.h[0];
      base[(fhi * 16 + nc + 1) * 16] = cv.h[1];
    }
  }
  async_wait_all();
  __syncthreads();

  v8f acc[4];
  zero_acc(acc);
  gemm_strip<64>(Q, 64, Kswz, wv, lane, acc);

  const float sc = 0.17677669529663688f;   // 1/sqrt(32)
  const int hi = (lane >> 4) & 1, nn = lane & 15;
#pragma unroll
  for (int nt = 0; nt < 4; ++nt)
#pragma unroll
    for (int v = 0; v < 8; ++v) {
      int i = wv * 16 + v + 8 * hi, l = nt * 16 + nn;
      scores_h[(((size_t)(m * 64 + i) * 64) + l) * 64 + k] = (_Float16)(acc[nt][v] * sc);
    }
}

// ---------------- stage C/D: f2 MLP, message = phi*weight, agg ------------
__global__ void __launch_bounds__(128)
k_weight_msg(const _Float16* __restrict__ scores_h, const _Float16* __restrict__ phi_h,
             const _Float16* __restrict__ wsw,
             const float* __restrict__ f2_b1, const float* __restrict__ f2_b2,
             _Float16* __restrict__ msg_h, float* __restrict__ agg) {
  __shared__ alignas(32) _Float16 S[64 * 64], W1[64 * 64], W2[64 * 64];
  __shared__ alignas(32) _Float16 stage[64 * 64], PH[64 * 64];
  __shared__ float MSG[64 * 64];
  const int tid = threadIdx.x, lane = tid & 31, wv = tid >> 5;
  const int bx = blockIdx.x;                 // m*64 + i
  const size_t slab = (size_t)bx * 4096;

  async_copy(S,  scores_h + slab,   512, tid, 128);
  async_copy(PH, phi_h + slab,      512, tid, 128);
  async_copy(W1, wsw + HOFF_F2W1,   512, tid, 128);
  async_copy(W2, wsw + HOFF_F2W2,   512, tid, 128);
  async_wait_all();
  __syncthreads();

  const int hi = (lane >> 4) & 1, nn = lane & 15;
  v8f acc[4];

  zero_acc(acc);
  gemm_strip<64>(S, 64, W1, wv, lane, acc);
#pragma unroll
  for (int nt = 0; nt < 4; ++nt) {
    float b = f2_b1[nt * 16 + nn];
#pragma unroll
    for (int v = 0; v < 8; ++v) {
      int row = wv * 16 + v + 8 * hi, col = nt * 16 + nn;
      stage[row * 64 + col] = (_Float16)tanhf(acc[nt][v] + b);
    }
  }
  __syncthreads();

  zero_acc(acc);
  gemm_strip<64>(stage, 64, W2, wv, lane, acc);
#pragma unroll
  for (int nt = 0; nt < 4; ++nt) {
    float b = f2_b2[nt * 16 + nn];
#pragma unroll
    for (int v = 0; v < 8; ++v) {
      int row = wv * 16 + v + 8 * hi, col = nt * 16 + nn;
      float w = acc[nt][v] + b;                       // "weight" tensor
      MSG[row * 64 + col] = w * (float)PH[row * 64 + col];
    }
  }
  __syncthreads();

  for (int i = tid; i < 4096; i += 128) msg_h[slab + i] = (_Float16)MSG[i];
  if (tid < 64) {                                    // agg over neighbor axis j
    float a = 0.0f;
    for (int j = 0; j < 64; ++j) a += MSG[j * 64 + tid];
    agg[(size_t)bx * 64 + tid] = a;
  }
}

// ---------------- stage E: edge update (g MLP, K=128) ---------------------
__global__ void __launch_bounds__(128)
k_edge_update(const _Float16* __restrict__ xij_h, const _Float16* __restrict__ msg_h,
              const _Float16* __restrict__ wsw,
              const float* __restrict__ g_b1, const float* __restrict__ g_b2,
              const float* __restrict__ xij0, float* __restrict__ out) {
  __shared__ alignas(32) _Float16 A[64 * 128], G1[128 * 64], G2[64 * 64];
  __shared__ alignas(32) _Float16 stage[64 * 64];
  const int tid = threadIdx.x, lane = tid & 31, wv = tid >> 5;
  const int bx = blockIdx.x;                 // m*64 + i
  const size_t slab = (size_t)bx * 4096;

  // interleave xij | msg into A rows via per-lane async LDS scatter
  {
    const char* xs = (const char*)(xij_h + slab);
    const char* ms = (const char*)(msg_h + slab);
    char* Ab = (char*)A;
    for (int i = tid; i < 512; i += 128) {
      int row = i >> 3, c = i & 7;             // 8 chunks (128B) per 64-half row
      async_cp16(Ab + (size_t)(row * 16 + c) * 16,     xs + (size_t)i * 16);
      async_cp16(Ab + (size_t)(row * 16 + 8 + c) * 16, ms + (size_t)i * 16);
    }
  }
  async_copy(G1, wsw + HOFF_GW1, 1024, tid, 128);
  async_copy(G2, wsw + HOFF_GW2,  512, tid, 128);
  async_wait_all();
  __syncthreads();

  const int hi = (lane >> 4) & 1, nn = lane & 15;
  v8f acc[4];

  zero_acc(acc);
  gemm_strip<128>(A, 128, G1, wv, lane, acc);
#pragma unroll
  for (int nt = 0; nt < 4; ++nt) {
    float b = g_b1[nt * 16 + nn];
#pragma unroll
    for (int v = 0; v < 8; ++v) {
      int row = wv * 16 + v + 8 * hi, col = nt * 16 + nn;
      stage[row * 64 + col] = (_Float16)tanhf(acc[nt][v] + b);
    }
  }
  __syncthreads();

  zero_acc(acc);
  gemm_strip<64>(stage, 64, G2, wv, lane, acc);
#pragma unroll
  for (int nt = 0; nt < 4; ++nt) {
    float b = g_b2[nt * 16 + nn];
#pragma unroll
    for (int v = 0; v < 8; ++v) {
      int j = wv * 16 + v + 8 * hi, col = nt * 16 + nn;
      out[OUT_EDGE_BASE + ((size_t)bx * 64 + j) * 96 + 32 + col] = acc[nt][v] + b;
    }
  }
  // prepend xij0 (32 features per edge)
  for (int idx = tid; idx < 2048; idx += 128) {
    int j = idx >> 5, c = idx & 31;
    out[OUT_EDGE_BASE + ((size_t)bx * 64 + j) * 96 + c] = xij0[(size_t)bx * 2048 + idx];
  }
}

// ---------------- stage F: node update (f MLP, K=128) ---------------------
__global__ void __launch_bounds__(128)
k_node_update(const float* __restrict__ x, const float* __restrict__ aggf,
              const _Float16* __restrict__ wsw,
              const float* __restrict__ f_b1, const float* __restrict__ f_b2,
              const float* __restrict__ x0, float* __restrict__ out) {
  __shared__ alignas(32) _Float16 A[64 * 128], F1[128 * 64], F2[64 * 64];
  __shared__ alignas(32) _Float16 stage[64 * 64];
  const int tid = threadIdx.x, lane = tid & 31, wv = tid >> 5;
  const int bx = blockIdx.x;                 // m
  const size_t slab = (size_t)bx * 4096;

  async_copy(F1, wsw + HOFF_FW1, 1024, tid, 128);
  async_copy(F2, wsw + HOFF_FW2,  512, tid, 128);
  for (int idx = tid; idx < 4096; idx += 128) {
    int row = idx >> 6, c = idx & 63;
    A[row * 128 + c]      = (_Float16)x[slab + idx];
    A[row * 128 + 64 + c] = (_Float16)aggf[slab + idx];
  }
  async_wait_all();
  __syncthreads();

  const int hi = (lane >> 4) & 1, nn = lane & 15;
  v8f acc[4];

  zero_acc(acc);
  gemm_strip<128>(A, 128, F1, wv, lane, acc);
#pragma unroll
  for (int nt = 0; nt < 4; ++nt) {
    float b = f_b1[nt * 16 + nn];
#pragma unroll
    for (int v = 0; v < 8; ++v) {
      int row = wv * 16 + v + 8 * hi, col = nt * 16 + nn;
      stage[row * 64 + col] = (_Float16)tanhf(acc[nt][v] + b);
    }
  }
  __syncthreads();

  zero_acc(acc);
  gemm_strip<64>(stage, 64, F2, wv, lane, acc);
#pragma unroll
  for (int nt = 0; nt < 4; ++nt) {
    float b = f_b2[nt * 16 + nn];
#pragma unroll
    for (int v = 0; v < 8; ++v) {
      int n = wv * 16 + v + 8 * hi, col = nt * 16 + nn;
      out[((size_t)(bx * 64 + n)) * 96 + 32 + col] = acc[nt][v] + b;
    }
  }
  for (int idx = tid; idx < 2048; idx += 128) {
    int n = idx >> 5, c = idx & 31;
    out[((size_t)(bx * 64 + n)) * 96 + c] = x0[(size_t)bx * 2048 + idx];
  }
}

// ---------------- launcher -------------------------------------------------
extern "C" void kernel_launch(void* const* d_in, const int* in_sizes, int n_in,
                              void* d_out, int out_size, void* d_ws, size_t ws_size,
                              hipStream_t stream) {
  (void)in_sizes; (void)n_in; (void)out_size; (void)ws_size;
  const float* x      = (const float*)d_in[0];
  const float* xij    = (const float*)d_in[1];
  const float* x0     = (const float*)d_in[2];
  const float* xij0   = (const float*)d_in[3];
  const float* wq     = (const float*)d_in[4];
  const float* wk     = (const float*)d_in[5];
  const float* phi_w1 = (const float*)d_in[6];
  const float* phi_b1 = (const float*)d_in[7];
  const float* phi_w2 = (const float*)d_in[8];
  const float* phi_b2 = (const float*)d_in[9];
  const float* f2_w1  = (const float*)d_in[10];
  const float* f2_b1  = (const float*)d_in[11];
  const float* f2_w2  = (const float*)d_in[12];
  const float* f2_b2  = (const float*)d_in[13];
  const float* f_w1   = (const float*)d_in[14];
  const float* f_b1   = (const float*)d_in[15];
  const float* f_w2   = (const float*)d_in[16];
  const float* f_b2   = (const float*)d_in[17];
  const float* g_w1   = (const float*)d_in[18];
  const float* g_b1   = (const float*)d_in[19];
  const float* g_w2   = (const float*)d_in[20];
  const float* g_b2   = (const float*)d_in[21];

  char* ws = (char*)d_ws;
  _Float16* wsw    = (_Float16*)ws;
  _Float16* xij_h  = (_Float16*)(ws + B_XIJH);
  _Float16* xqT    = (_Float16*)(ws + B_XQT);
  _Float16* xkT    = (_Float16*)(ws + B_XKT);
  _Float16* phi_h  = (_Float16*)(ws + B_PHI);
  _Float16* sco_h  = (_Float16*)(ws + B_SCO);
  _Float16* msg_h  = (_Float16*)(ws + B_MSG);
  float*    agg    = (float*)(ws + B_AGG);
  float*    out    = (float*)d_out;

  prep_weights<<<10, 256, 0, stream>>>(wq, wk, phi_w1, phi_w2, f2_w1, f2_w2,
                                       f_w1, f_w2, g_w1, g_w2, wsw);
  k_edge_pre<<<MD * ND, 128, 0, stream>>>(xij, wsw, phi_b1, phi_b2,
                                          xij_h, xqT, xkT, phi_h);
  k_scores<<<MD * DD, 128, 0, stream>>>(xqT, xkT, sco_h);
  k_weight_msg<<<MD * ND, 128, 0, stream>>>(sco_h, phi_h, wsw, f2_b1, f2_b2,
                                            msg_h, agg);
  k_edge_update<<<MD * ND, 128, 0, stream>>>(xij_h, msg_h, wsw, g_b1, g_b2,
                                             xij0, out);
  k_node_update<<<MD, 128, 0, stream>>>(x, agg, wsw, f_b1, f_b2, x0, out);
}
